// AttentionBlock_65678639891143
// MI455X (gfx1250) — matrix operand order repaired and checked by reading, not compiled, not run on hardware
//
#include <hip/hip_runtime.h>

#define BS   8
#define CIN  128
#define CH   64
#define NTOK 4096

typedef __attribute__((ext_vector_type(16))) __bf16 v16bf;
typedef __attribute__((ext_vector_type(8)))  float  v8f;
typedef unsigned short u16;
typedef unsigned int   u32;
typedef unsigned long long u64;

__device__ __forceinline__ u16 f2bf_bits(float f) {
  union { __bf16 h; u16 s; } v; v.h = (__bf16)f;
  return v.s;
}
__device__ __forceinline__ u32 pack_bf16(float lo, float hi) {
  union { __bf16 h[2]; u32 u; } v;
  v.h[0] = (__bf16)lo; v.h[1] = (__bf16)hi;
  return v.u;
}

// B fragment (K x 16) from LDS stored col-major: Xt[col][k], contiguous k.
// lane -> col = subcol0 + (lane&15); element s -> K = kbase + (lane>>4)*8 + s (s<8), kbase + 16 + (lane>>4)*8 + (s-8) (s>=8)
__device__ __forceinline__ v16bf ldsB(const u16* Xt, int stride, int subcol0, int kbase) {
  int lane = threadIdx.x & 31;
  int col  = subcol0 + (lane & 15);
  int k0   = kbase + ((lane >> 4) << 3);
  const u32* p = (const u32*)(Xt + col * stride + k0);
  union { v16bf v; u32 u[8]; } r;
#pragma unroll
  for (int i = 0; i < 4; ++i) { r.u[i] = p[i]; r.u[4 + i] = p[8 + i]; }
  return r.v;
}

// A fragment (16 x 32) from LDS row-major bf16: P[row][k].
__device__ __forceinline__ v16bf ldsA(const u16* P, int stride, int kb) {
  int lane = threadIdx.x & 31;
  int row  = lane & 15;
  int kg   = (lane >> 4) << 3;
  const u16* p  = P + row * stride + kb;
  const u32* p0 = (const u32*)(p + kg);
  const u32* p1 = (const u32*)(p + 16 + kg);
  union { v16bf v; u32 u[8]; } r;
#pragma unroll
  for (int i = 0; i < 4; ++i) r.u[i] = p0[i];
#pragma unroll
  for (int i = 0; i < 4; ++i) r.u[4 + i] = p1[i];
  return r.v;
}

// A fragment (16 x 32) from global fp32 weights W[row][k], converted to bf16
__device__ __forceinline__ v16bf gldA_w(const float* Wg, int ld, int m0, int kb) {
  int lane = threadIdx.x & 31;
  int row  = m0 + (lane & 15);
  int kg   = (lane >> 4) << 3;
  const float* p = Wg + row * ld + kb;
  union { v16bf v; u32 u[8]; } r;
#pragma unroll
  for (int i = 0; i < 4; ++i) r.u[i] = pack_bf16(p[kg + 2 * i], p[kg + 2 * i + 1]);
#pragma unroll
  for (int i = 0; i < 4; ++i) r.u[4 + i] = pack_bf16(p[16 + kg + 2 * i], p[16 + kg + 2 * i + 1]);
  return r.v;
}

// A fragment (16 x 32) from global bf16 token-major matrix Q[row][k]
__device__ __forceinline__ v16bf gldA_bf(const u16* Qg, int ld, int row0, int kb) {
  int lane = threadIdx.x & 31;
  int row  = row0 + (lane & 15);
  int kg   = (lane >> 4) << 3;
  const u16* p  = Qg + (size_t)row * ld + kb;
  const u32* p0 = (const u32*)(p + kg);
  const u32* p1 = (const u32*)(p + 16 + kg);
  union { v16bf v; u32 u[8]; } r;
#pragma unroll
  for (int i = 0; i < 4; ++i) r.u[i] = p0[i];
#pragma unroll
  for (int i = 0; i < 4; ++i) r.u[4 + i] = p1[i];
  return r.v;
}

__device__ __forceinline__ v8f wmma_bf16(v16bf a, v16bf b, v8f c) {
  return __builtin_amdgcn_wmma_f32_16x16x32_bf16(false, a, false, b, (short)0, c, false, false);
}

// ---------------------------------------------------------------------------
// Projection kernel: Q = relu(W_tgt*tgt)  -> [b][tok][64] bf16
//                    K = relu(W_ref*ref)  -> [b][tok][64] bf16
//                    V = bf16(ref)        -> [b][c][n]   bf16
//                    G = W_out*tgt + b    -> [b][c][n]   fp32
// ---------------------------------------------------------------------------
__launch_bounds__(128)
__global__ void proj_kernel(const float* __restrict__ tgt, const float* __restrict__ ref,
                            const float* __restrict__ Wtgt, const float* __restrict__ btgt,
                            const float* __restrict__ Wref, const float* __restrict__ bref,
                            const float* __restrict__ Wout, const float* __restrict__ bout,
                            u16* __restrict__ Qb, u16* __restrict__ Kb,
                            u16* __restrict__ Vb, float* __restrict__ Gb) {
  const int b   = blockIdx.y;
  const int t0  = blockIdx.x * 64;
  const int tid = threadIdx.x;
  const int TS  = 136;                      // padded stride

  __shared__ u16 s_tgt[64 * 136];           // [token][c] bf16
  __shared__ u16 s_ref[64 * 136];

  // fill LDS (packing channel pairs into u32 writes)
  for (int i = tid; i < 64 * 64; i += 128) {
    int j = i & 63, c = (i >> 6) << 1;
    size_t gi = ((size_t)(b * CIN + c)) * NTOK + t0 + j;
    *(u32*)&s_tgt[j * TS + c] = pack_bf16(tgt[gi], tgt[gi + NTOK]);
    *(u32*)&s_ref[j * TS + c] = pack_bf16(ref[gi], ref[gi + NTOK]);
  }
  // V = bf16(ref), [c][n] layout, coalesced paired stores
  for (int i = tid; i < CIN * 32; i += 128) {
    int c = i >> 5, j2 = (i & 31) << 1;
    size_t gi = ((size_t)(b * CIN + c)) * NTOK + t0 + j2;
    *(u32*)&Vb[gi] = pack_bf16(ref[gi], ref[gi + 1]);
  }
  __syncthreads();

  const int wv   = tid >> 5;
  const int lane = tid & 31;
  const int col  = lane & 15;
  const int rsel = (lane >> 4) << 3;
  const int tok  = t0 + wv * 16 + col;

  v16bf Btg[4], Brf[4];
#pragma unroll
  for (int kf = 0; kf < 4; ++kf) {
    Btg[kf] = ldsB(s_tgt, TS, wv * 16, kf * 32);
    Brf[kf] = ldsB(s_ref, TS, wv * 16, kf * 32);
  }

  const v8f vzero = {};

  // Q projection: 4 M-tiles, store transposed [tok][64], paired u32 stores
#pragma unroll
  for (int mt = 0; mt < 4; ++mt) {
    v8f acc = vzero;
#pragma unroll
    for (int kf = 0; kf < 4; ++kf)
      acc = wmma_bf16(gldA_w(Wtgt, CIN, mt * 16, kf * 32), Btg[kf], acc);
    int row0 = mt * 16 + rsel;
    size_t base = ((size_t)(b * NTOK + tok)) * CH + row0;
#pragma unroll
    for (int r2 = 0; r2 < 4; ++r2) {
      float v0 = fmaxf(acc[2 * r2]     + btgt[row0 + 2 * r2], 0.f);
      float v1 = fmaxf(acc[2 * r2 + 1] + btgt[row0 + 2 * r2 + 1], 0.f);
      *(u32*)&Qb[base + 2 * r2] = pack_bf16(v0, v1);
    }
  }
  // K projection, same token-major layout
#pragma unroll
  for (int mt = 0; mt < 4; ++mt) {
    v8f acc = vzero;
#pragma unroll
    for (int kf = 0; kf < 4; ++kf)
      acc = wmma_bf16(gldA_w(Wref, CIN, mt * 16, kf * 32), Brf[kf], acc);
    int row0 = mt * 16 + rsel;
    size_t base = ((size_t)(b * NTOK + tok)) * CH + row0;
#pragma unroll
    for (int r2 = 0; r2 < 4; ++r2) {
      float v0 = fmaxf(acc[2 * r2]     + bref[row0 + 2 * r2], 0.f);
      float v1 = fmaxf(acc[2 * r2 + 1] + bref[row0 + 2 * r2 + 1], 0.f);
      *(u32*)&Kb[base + 2 * r2] = pack_bf16(v0, v1);
    }
  }
  // Gate projection: 8 M-tiles, fp32 [c][n]
#pragma unroll
  for (int mt = 0; mt < 8; ++mt) {
    v8f acc = vzero;
#pragma unroll
    for (int kf = 0; kf < 4; ++kf)
      acc = wmma_bf16(gldA_w(Wout, CIN, mt * 16, kf * 32), Btg[kf], acc);
#pragma unroll
    for (int r = 0; r < 8; ++r) {
      int row = mt * 16 + rsel + r;
      Gb[((size_t)(b * CIN + row)) * NTOK + tok] = acc[r] + bout[row];
    }
  }
}

// ---------------------------------------------------------------------------
// Flash-attention kernel: 256 threads (8 waves), wave handles 16 query rows.
// 64-wide key tiles; double-buffered async global->LDS DMA for K and V.
// ---------------------------------------------------------------------------
__launch_bounds__(256)
__global__ void flash_kernel(const u16* __restrict__ Qb,
                             const u16* __restrict__ Kb,
                             const u16* __restrict__ Vb,
                             const float* __restrict__ Gb,
                             float* __restrict__ out) {
  const int b    = blockIdx.y;
  const int t0   = blockIdx.x * 128;
  const int tid  = threadIdx.x;
  const int wv   = tid >> 5;
  const int lane = tid & 31;
  const int col  = lane & 15;
  const int rsel = (lane >> 4) << 3;

  const int KSTR = 72;   // sKt: [m 64][k 64] padded (144B rows, 16B aligned)
  const int VSTR = 72;   // sVt: [c 128][m 64] padded
  const int PSTR = 68;   // sP : [row 16][k 64] padded

  __shared__ u16 sKt[2][64 * 72];
  __shared__ u16 sVt[2][128 * 72];
  __shared__ u16 sP[8][16 * 68];

  const int r0 = t0 + wv * 16;
  const u16* Qbb = Qb + (size_t)b * NTOK * CH;
  const u16* Kbb = Kb + (size_t)b * NTOK * CH;
  const u16* Vbb = Vb + (size_t)b * CIN * NTOK;

  // issue async DMA for K tile (8KB) + V tile (16KB) of key block m0 into buffer `buf`
  auto issue_tile = [&](int buf, int m0) {
    u64 kbase = (u64)(Kbb + (size_t)m0 * CH);
#pragma unroll
    for (int c2 = 0; c2 < 2; ++c2) {
      int chunk = tid + c2 * 256;           // 512 x 16B chunks
      int row = chunk >> 3, seg = chunk & 7;
      u32 ldsoff = (u32)(uintptr_t)(&sKt[buf][row * KSTR + seg * 8]);
      u32 goff   = (u32)(row * 128 + seg * 16);
      asm volatile("global_load_async_to_lds_b128 %0, %1, %2 offset:0"
                   :: "v"(ldsoff), "v"(goff), "s"(kbase) : "memory");
    }
    u64 vbase = (u64)(Vbb + m0);            // row stride NTOK*2 bytes
#pragma unroll
    for (int c2 = 0; c2 < 4; ++c2) {
      int chunk = tid + c2 * 256;           // 1024 x 16B chunks
      int row = chunk >> 3, seg = chunk & 7;
      u32 ldsoff = (u32)(uintptr_t)(&sVt[buf][row * VSTR + seg * 8]);
      u32 goff   = (u32)(row * (NTOK * 2) + seg * 16);
      asm volatile("global_load_async_to_lds_b128 %0, %1, %2 offset:0"
                   :: "v"(ldsoff), "v"(goff), "s"(vbase) : "memory");
    }
  };

  // Q fragments live in registers for the whole loop
  v16bf qa0 = gldA_bf(Qbb, CH, r0, 0);
  v16bf qa1 = gldA_bf(Qbb, CH, r0, 32);

  const v8f vzero = {};
  v8f   o[8];
  float mrow[8], lrow[8];
#pragma unroll
  for (int i = 0; i < 8; ++i) { o[i] = vzero; mrow[i] = -1e30f; lrow[i] = 0.f; }

  issue_tile(0, 0);
  asm volatile("s_wait_asynccnt 0x0" ::: "memory");
  __syncthreads();

  const int NT = NTOK / 64;
  for (int it = 0; it < NT; ++it) {
    const int cur = it & 1;
    if (it + 1 < NT) issue_tile(cur ^ 1, (it + 1) * 64);  // DMA next under compute

    const u16* Kt = sKt[cur];
    const u16* Vt = sVt[cur];

    // --- S = Q * K^T : 16 rows x 64 cols in four C-fragments
    v8f s[4];
#pragma unroll
    for (int ct = 0; ct < 4; ++ct) {
      s[ct] = vzero;
      s[ct] = wmma_bf16(qa0, ldsB(Kt, KSTR, ct * 16, 0),  s[ct]);
      s[ct] = wmma_bf16(qa1, ldsB(Kt, KSTR, ct * 16, 32), s[ct]);
    }

    // --- online softmax (lanes 0-15 = row r, lanes 16-31 = row r+8)
#pragma unroll
    for (int r = 0; r < 8; ++r) {
      float tm = fmaxf(fmaxf(s[0][r], s[1][r]), fmaxf(s[2][r], s[3][r]));
#pragma unroll
      for (int x = 1; x < 16; x <<= 1) tm = fmaxf(tm, __shfl_xor(tm, x, 32));
      float mn = fmaxf(mrow[r], tm);
      float sc = __expf(mrow[r] - mn);
      float p0 = __expf(s[0][r] - mn);
      float p1 = __expf(s[1][r] - mn);
      float p2 = __expf(s[2][r] - mn);
      float p3 = __expf(s[3][r] - mn);
      float ts = (p0 + p1) + (p2 + p3);
#pragma unroll
      for (int x = 1; x < 16; x <<= 1) ts += __shfl_xor(ts, x, 32);
      mrow[r] = mn;
      lrow[r] = lrow[r] * sc + ts;
#pragma unroll
      for (int ct = 0; ct < 8; ++ct) o[ct][r] *= sc;
      int pb = (rsel + r) * PSTR + col;
      sP[wv][pb]      = f2bf_bits(p0);
      sP[wv][pb + 16] = f2bf_bits(p1);
      sP[wv][pb + 32] = f2bf_bits(p2);
      sP[wv][pb + 48] = f2bf_bits(p3);
    }

    // --- O += P * V  (K = 64 in two 32-deep steps, eight 16-wide c tiles)
    v16bf ap0 = ldsA(sP[wv], PSTR, 0);
    v16bf ap1 = ldsA(sP[wv], PSTR, 32);
#pragma unroll
    for (int ct = 0; ct < 8; ++ct) {
      o[ct] = wmma_bf16(ap0, ldsB(Vt, VSTR, ct * 16, 0),  o[ct]);
      o[ct] = wmma_bf16(ap1, ldsB(Vt, VSTR, ct * 16, 32), o[ct]);
    }

    asm volatile("s_wait_asynccnt 0x0" ::: "memory");
    __syncthreads();
  }

  // --- epilogue: normalize, apply gate, write [b][c][n]
  const float* Gbb  = Gb  + (size_t)b * CIN * NTOK;
  float*       outb = out + (size_t)b * CIN * NTOK;
#pragma unroll
  for (int r = 0; r < 8; ++r) {
    int   row = r0 + rsel + r;
    float inv = 1.0f / lrow[r];
#pragma unroll
    for (int ct = 0; ct < 8; ++ct) {
      int c = ct * 16 + col;
      size_t idx = (size_t)c * NTOK + row;
      outb[idx] = o[ct][r] * inv * Gbb[idx];
    }
  }
}

extern "C" void kernel_launch(void* const* d_in, const int* in_sizes, int n_in,
                              void* d_out, int out_size, void* d_ws, size_t ws_size,
                              hipStream_t stream) {
  (void)in_sizes; (void)n_in; (void)out_size; (void)ws_size;
  const float* tgt  = (const float*)d_in[0];
  const float* ref  = (const float*)d_in[1];
  const float* Wtgt = (const float*)d_in[2];
  const float* btgt = (const float*)d_in[3];
  const float* Wref = (const float*)d_in[4];
  const float* bref = (const float*)d_in[5];
  const float* Wout = (const float*)d_in[6];
  const float* bout = (const float*)d_in[7];
  float* out = (float*)d_out;

  // workspace: Q bf16 (4MB) | K bf16 (4MB) | V bf16 (8MB) | G fp32 (16MB) = 32MB
  u16*   Qb = (u16*)d_ws;
  u16*   Kb = (u16*)((char*)d_ws + ((size_t)4  << 20));
  u16*   Vb = (u16*)((char*)d_ws + ((size_t)8  << 20));
  float* Gb = (float*)((char*)d_ws + ((size_t)16 << 20));

  dim3 pg(NTOK / 64, BS);
  proj_kernel<<<pg, 128, 0, stream>>>(tgt, ref, Wtgt, btgt, Wref, bref, Wout, bout,
                                      Qb, Kb, Vb, Gb);
  dim3 fg(NTOK / 128, BS);
  flash_kernel<<<fg, 256, 0, stream>>>(Qb, Kb, Vb, Gb, out);
}
